// STGCN_MLP_60902636257629
// MI455X (gfx1250) — matrix operand order repaired and checked by reading, not compiled, not run on hardware
//
#include <hip/hip_runtime.h>
#include <hip/hip_bf16.h>

typedef __attribute__((ext_vector_type(16))) __bf16 v16bf;
typedef __attribute__((ext_vector_type(8)))  float  v8f;

#define N_NODES 14
#define NPAD    16
#define SEQ     24
#define KP      32   // padded K for SEQ
#define GH      64   // GAT hidden
#define MH      64   // MLP hidden
#define BATCH   32768
#define NEDGE   56
#define ETOT    70   // 56 edges + 14 self loops
#define SPB     16   // samples per block
#define XBYTES  (SPB * SEQ * N_NODES * 4)   // 21504 B, contiguous per block
#define XVEC    (XBYTES / 16)               // 1344 b128 lane-transfers
#define OBYTES  (N_NODES * SPB * SEQ * 4)   // 21504 B out staging
#define OVEC    (OBYTES / 16)               // 1344

// ---- WMMA fragment helpers (layouts per CDNA5 ISA 7.12.2) ----

// A-matrix 16x32 bf16: lane L -> row m=L&15, half hi=L>>4.
// VGPR j(0..3): K = hi*8 + 2j, +1 ; VGPR j(4..7): K = 16 + hi*8 + 2(j-4), +1
__device__ inline v16bf load_a_frag(const __bf16* p, int rowStride, int kOff) {
    const int lane = threadIdx.x & 31;
    const int m = lane & 15, hi = lane >> 4;
    const __bf16* r = p + m * rowStride + kOff;
    v16bf a;
#pragma unroll
    for (int j = 0; j < 4; ++j) {
        int k0 = hi * 8 + 2 * j;
        a[2 * j]     = r[k0];
        a[2 * j + 1] = r[k0 + 1];
        int k1 = 16 + hi * 8 + 2 * j;
        a[2 * (j + 4)]     = r[k1];
        a[2 * (j + 4) + 1] = r[k1 + 1];
    }
    return a;
}

// B-matrix 32x16 bf16 from global fp32 weight W[k][n] (row-major, stride nStride):
// lane L -> col n=L&15, half hi=L>>4 ; VGPR j(0..3): K = hi*8 + 2j, +1 ; VGPR j(4..7): K = 16 + hi*8 + 2(j-4), +1
__device__ inline v16bf load_b_frag(const float* W, int nStride, int kOff, int kLim,
                                    int cOff, int nLim) {
    const int lane = threadIdx.x & 31;
    const int n = lane & 15, hi = lane >> 4;
    const int col = cOff + n;
    v16bf b;
#pragma unroll
    for (int j = 0; j < 8; ++j) {
        int k = kOff + (j < 4 ? hi * 8 + 2 * j : 16 + hi * 8 + 2 * (j - 4));
        float f0 = (col < nLim && k     < kLim) ? W[(size_t)k       * nStride + col] : 0.f;
        float f1 = (col < nLim && k + 1 < kLim) ? W[(size_t)(k + 1) * nStride + col] : 0.f;
        b[2 * j]     = (__bf16)f0;
        b[2 * j + 1] = (__bf16)f1;
    }
    return b;
}

// C/D 16x16 f32: VGPR r, lane L -> row m = r + (L>>4)*8, col n = L&15
__device__ inline void store_c_lds(__bf16* p, int rowStride, int cOff, const v8f& c) {
    const int lane = threadIdx.x & 31;
    const int n = lane & 15, hi = lane >> 4;
#pragma unroll
    for (int r = 0; r < 8; ++r) {
        int m = r + hi * 8;
        p[m * rowStride + cOff + n] = (__bf16)c[r];
    }
}

__global__ __launch_bounds__(256, 1) void stgcn_fused_kernel(
    const float* __restrict__ x, const int* __restrict__ ei,
    const float* __restrict__ W1, const float* __restrict__ as1,
    const float* __restrict__ ad1, const float* __restrict__ b1,
    const float* __restrict__ W2, const float* __restrict__ as2,
    const float* __restrict__ ad2, const float* __restrict__ b2,
    const float* __restrict__ f1w, const float* __restrict__ f1b,
    const float* __restrict__ f2w, const float* __restrict__ f2b,
    float* __restrict__ out)
{
    __shared__ __align__(16) __bf16 sA[SPB * NPAD * GH];   // 32 KB ping buffer
    __shared__ __align__(16) __bf16 sB[SPB * NPAD * GH];   // 32 KB pong buffer
    __shared__ float  sP[SPB][NPAD];
    __shared__ float  sQ[SPB][NPAD];
    __shared__ int    sAdjN[NPAD];
    __shared__ unsigned char sAdjS[NPAD][ETOT + 2];

    const int t    = threadIdx.x;
    const int wave = t >> 5;
    const int s0   = blockIdx.x * SPB;

    float* sXf  = (float*)sB;   // stage-0 raw x staging (21504 B, dies before GEMM1)
    float* sOut = (float*)sA;   // output staging [14][16][24] fp32 (lives after fc1)

    // ---- kick off async DMA of the contiguous x tile: global -> LDS ----
    {
        const char*  gbase  = (const char*)(x + (size_t)s0 * SEQ * N_NODES);
        unsigned     lbase  = (unsigned)(size_t)(void*)sXf;
        for (int idx = t; idx < XVEC; idx += 256) {
            unsigned long long gaddr = (unsigned long long)(size_t)(gbase + idx * 16);
            unsigned           laddr = lbase + idx * 16;
            asm volatile("global_load_async_to_lds_b128 %0, %1, off"
                         :: "v"(laddr), "v"(gaddr) : "memory");
        }
    }

    // ---- build per-dst adjacency (graph identical for all samples) ----
    if (t == 0) {
        for (int d = 0; d < NPAD; ++d) sAdjN[d] = 0;
        for (int e = 0; e < ETOT; ++e) {
            int sr = (e < NEDGE) ? ei[e]         : (e - NEDGE);
            int ds = (e < NEDGE) ? ei[NEDGE + e] : (e - NEDGE);
            sAdjS[ds][sAdjN[ds]++] = (unsigned char)sr;
        }
    }

    // ---- zero-fill padded A tile while the DMA is in flight ----
    for (int i = t; i < SPB * NPAD * KP; i += 256) sA[i] = (__bf16)0.f;

    asm volatile("s_wait_asynccnt 0x0" ::: "memory");
    __syncthreads();

    // ---- transpose/convert: sXf[s][k][n] fp32 -> sA h0[s][n][k] bf16 ----
    for (int i = t; i < SPB * SEQ * N_NODES; i += 256) {
        int n = i % N_NODES;
        int k = (i / N_NODES) % SEQ;
        int s = i / (N_NODES * SEQ);
        sA[(s * NPAD + n) * KP + k] = (__bf16)sXf[i];   // i is linear in sXf
    }
    __syncthreads();

    // ---- stage 1: hlin1 = h0 @ W1  -> sB [16][16][64] ----
    {
        v16bf bw[4];
#pragma unroll
        for (int nt = 0; nt < 4; ++nt) bw[nt] = load_b_frag(W1, GH, 0, SEQ, nt * 16, GH);
#pragma unroll
        for (int rep = 0; rep < 2; ++rep) {
            int s = wave + rep * 8;
            v16bf a = load_a_frag(sA + s * NPAD * KP, KP, 0);
#pragma unroll
            for (int nt = 0; nt < 4; ++nt) {
                v8f c = {};
                c = __builtin_amdgcn_wmma_f32_16x16x32_bf16(false, a, false, bw[nt],
                                                            (short)0, c, false, false);
                store_c_lds(sB + s * NPAD * GH, GH, nt * 16, c);
            }
        }
    }
    __syncthreads();

    // ---- attention scores p,q for layer 1 ----
    {
        int s = t >> 4, n = t & 15;
        const __bf16* h = sB + (s * NPAD + n) * GH;
        float p = 0.f, q = 0.f;
        for (int f = 0; f < GH; ++f) {
            float hv = (float)h[f];
            p += hv * as1[f];
            q += hv * ad1[f];
        }
        sP[s][n] = p; sQ[s][n] = q;
    }
    __syncthreads();

    // ---- edge softmax + aggregate + bias + ELU -> sA h1[s][n][64] ----
    {
        int s = t >> 4, d = t & 15;
        if (d < N_NODES) {
            int cnt = sAdjN[d];
            float qd = sQ[s][d];
            float mx = -1e30f;
            for (int i = 0; i < cnt; ++i) {
                float a = sP[s][sAdjS[d][i]] + qd;
                a = a > 0.f ? a : a * 0.2f;
                mx = fmaxf(mx, a);
            }
            float sum = 0.f;
            for (int i = 0; i < cnt; ++i) {
                float a = sP[s][sAdjS[d][i]] + qd;
                a = a > 0.f ? a : a * 0.2f;
                sum += __expf(a - mx);
            }
            float inv = 1.f / sum;
            const __bf16* hb = sB + s * NPAD * GH;
            __bf16* ob = sA + (s * NPAD + d) * GH;
            for (int fb = 0; fb < GH; fb += 16) {
                float acc[16];
#pragma unroll
                for (int j = 0; j < 16; ++j) acc[j] = 0.f;
                for (int i = 0; i < cnt; ++i) {
                    int sr = sAdjS[d][i];
                    float a = sP[s][sr] + qd;
                    a = a > 0.f ? a : a * 0.2f;
                    float w = __expf(a - mx) * inv;
                    const __bf16* hr = hb + sr * GH + fb;
#pragma unroll
                    for (int j = 0; j < 16; ++j) acc[j] += w * (float)hr[j];
                }
#pragma unroll
                for (int j = 0; j < 16; ++j) {
                    float v = acc[j] + b1[fb + j];
                    v = v > 0.f ? v : (__expf(v) - 1.f);   // ELU
                    ob[fb + j] = (__bf16)v;
                }
            }
        }
    }
    __syncthreads();

    // ---- stage 2: hlin2 = h1 @ W2 -> sB [16][16][32] (cols 24..31 = 0) ----
    {
        v16bf bw[2][2];
#pragma unroll
        for (int kt = 0; kt < 2; ++kt)
#pragma unroll
            for (int nt = 0; nt < 2; ++nt)
                bw[kt][nt] = load_b_frag(W2, SEQ, kt * 32, GH, nt * 16, SEQ);
#pragma unroll
        for (int rep = 0; rep < 2; ++rep) {
            int s = wave + rep * 8;
            v16bf a0 = load_a_frag(sA + s * NPAD * GH, GH, 0);
            v16bf a1 = load_a_frag(sA + s * NPAD * GH, GH, 32);
#pragma unroll
            for (int nt = 0; nt < 2; ++nt) {
                v8f c = {};
                c = __builtin_amdgcn_wmma_f32_16x16x32_bf16(false, a0, false, bw[0][nt],
                                                            (short)0, c, false, false);
                c = __builtin_amdgcn_wmma_f32_16x16x32_bf16(false, a1, false, bw[1][nt],
                                                            (short)0, c, false, false);
                store_c_lds(sB + s * NPAD * KP, KP, nt * 16, c);
            }
        }
    }
    __syncthreads();

    // ---- attention scores p,q for layer 2 ----
    {
        int s = t >> 4, n = t & 15;
        const __bf16* h = sB + (s * NPAD + n) * KP;
        float p = 0.f, q = 0.f;
        for (int f = 0; f < SEQ; ++f) {
            float hv = (float)h[f];
            p += hv * as2[f];
            q += hv * ad2[f];
        }
        sP[s][n] = p; sQ[s][n] = q;
    }
    __syncthreads();

    // ---- edge softmax + aggregate + bias -> sA g[s][n][32] (k-pad zero) ----
    {
        int s = t >> 4, d = t & 15;
        if (d < N_NODES) {
            int cnt = sAdjN[d];
            float qd = sQ[s][d];
            float mx = -1e30f;
            for (int i = 0; i < cnt; ++i) {
                float a = sP[s][sAdjS[d][i]] + qd;
                a = a > 0.f ? a : a * 0.2f;
                mx = fmaxf(mx, a);
            }
            float sum = 0.f;
            for (int i = 0; i < cnt; ++i) {
                float a = sP[s][sAdjS[d][i]] + qd;
                a = a > 0.f ? a : a * 0.2f;
                sum += __expf(a - mx);
            }
            float inv = 1.f / sum;
            const __bf16* hb = sB + s * NPAD * KP;
            __bf16* ob = sA + (s * NPAD + d) * KP;
            for (int fb = 0; fb < KP; fb += 16) {
                float acc[16];
#pragma unroll
                for (int j = 0; j < 16; ++j) acc[j] = 0.f;
                for (int i = 0; i < cnt; ++i) {
                    int sr = sAdjS[d][i];
                    float a = sP[s][sr] + qd;
                    a = a > 0.f ? a : a * 0.2f;
                    float w = __expf(a - mx) * inv;
                    const __bf16* hr = hb + sr * KP + fb;
#pragma unroll
                    for (int j = 0; j < 16; ++j) acc[j] += w * (float)hr[j];
                }
#pragma unroll
                for (int j = 0; j < 16; ++j) {
                    int f = fb + j;
                    float v = (f < SEQ) ? (acc[j] + b2[f]) : 0.f;
                    ob[f] = (__bf16)v;   // no activation between GAT2 and MLP
                }
            }
        }
    }
    __syncthreads();

    // ---- per-node MLP phase 1: fc1 + relu -> sB hid[(wave*2+rep)][16][64] ----
    {
        const int lane = t & 31;
        const int ncol = lane & 15, hi = lane >> 4;
#pragma unroll
        for (int rep = 0; rep < 2; ++rep) {
            int n = wave + rep * 8;
            if (n < N_NODES) {
                __bf16* hid = sB + (wave * 2 + rep) * (16 * MH);
                v16bf a = load_a_frag(sA + n * KP, NPAD * KP, 0);
                const float* w1n = f1w + (size_t)n * SEQ * MH;
#pragma unroll
                for (int nt = 0; nt < 4; ++nt) {
                    v16bf b = load_b_frag(w1n, MH, 0, SEQ, nt * 16, MH);
                    v8f c = {};
                    c = __builtin_amdgcn_wmma_f32_16x16x32_bf16(false, a, false, b,
                                                                (short)0, c, false, false);
#pragma unroll
                    for (int r = 0; r < 8; ++r) {
                        int m = r + hi * 8;
                        float v = c[r] + f1b[n * MH + nt * 16 + ncol];
                        v = v > 0.f ? v : 0.f;
                        hid[m * MH + nt * 16 + ncol] = (__bf16)v;
                    }
                }
            }
        }
    }
    __syncthreads();   // g (sA) dead from here; hid (sB) complete

    // ---- phase 2: fc2 + bias -> sOut[n][16][24] fp32 (aliases sA) ----
    {
        const int lane = t & 31;
        const int ncol = lane & 15, hi = lane >> 4;
#pragma unroll
        for (int rep = 0; rep < 2; ++rep) {
            int n = wave + rep * 8;
            if (n < N_NODES) {
                const __bf16* hid = sB + (wave * 2 + rep) * (16 * MH);
                v16bf a0 = load_a_frag(hid, MH, 0);
                v16bf a1 = load_a_frag(hid, MH, 32);
                const float* w2n = f2w + (size_t)n * MH * SEQ;
#pragma unroll
                for (int nt = 0; nt < 2; ++nt) {
                    v16bf bk0 = load_b_frag(w2n, SEQ, 0,  MH, nt * 16, SEQ);
                    v16bf bk1 = load_b_frag(w2n, SEQ, 32, MH, nt * 16, SEQ);
                    v8f c = {};
                    c = __builtin_amdgcn_wmma_f32_16x16x32_bf16(false, a0, false, bk0,
                                                                (short)0, c, false, false);
                    c = __builtin_amdgcn_wmma_f32_16x16x32_bf16(false, a1, false, bk1,
                                                                (short)0, c, false, false);
                    int col = nt * 16 + ncol;
                    if (col < SEQ) {
#pragma unroll
                        for (int r = 0; r < 8; ++r) {
                            int m = r + hi * 8;
                            sOut[(n * SPB + m) * SEQ + col] = c[r] + f2b[n * SEQ + col];
                        }
                    }
                }
            }
        }
    }
    __syncthreads();   // sOut complete and visible

    // ---- drain: async store LDS -> global (layout matches per-node chunks) ----
    {
        unsigned lbase = (unsigned)(size_t)(void*)sOut;
        for (int q = t; q < OVEC; q += 256) {
            int n   = q / (SPB * SEQ / 4);        // 96 float4 per node
            int rem = q % (SPB * SEQ / 4);
            const char* gp = (const char*)(out + (size_t)n * BATCH * SEQ
                                               + (size_t)s0 * SEQ) + rem * 16;
            unsigned long long gaddr = (unsigned long long)(size_t)gp;
            unsigned           laddr = lbase + q * 16;
            asm volatile("global_store_async_from_lds_b128 %0, %1, off"
                         :: "v"(gaddr), "v"(laddr) : "memory");
        }
    }
    asm volatile("s_wait_asynccnt 0x0" ::: "memory");
}

extern "C" void kernel_launch(void* const* d_in, const int* in_sizes, int n_in,
                              void* d_out, int out_size, void* d_ws, size_t ws_size,
                              hipStream_t stream) {
    const float* x   = (const float*)d_in[0];
    const int*   ei  = (const int*)  d_in[1];
    const float* W1  = (const float*)d_in[2];
    const float* as1 = (const float*)d_in[3];
    const float* ad1 = (const float*)d_in[4];
    const float* b1  = (const float*)d_in[5];
    const float* W2  = (const float*)d_in[6];
    const float* as2 = (const float*)d_in[7];
    const float* ad2 = (const float*)d_in[8];
    const float* b2  = (const float*)d_in[9];
    const float* f1w = (const float*)d_in[10];
    const float* f1b = (const float*)d_in[11];
    const float* f2w = (const float*)d_in[12];
    const float* f2b = (const float*)d_in[13];
    float* out = (float*)d_out;

    dim3 grid(BATCH / SPB);
    dim3 block(256);
    stgcn_fused_kernel<<<grid, block, 0, stream>>>(
        x, ei, W1, as1, ad1, b1, W2, as2, ad2, b2, f1w, f1b, f2w, f2b, out);
}
